// AliNetGraphAttentionLayer_5798205850097
// MI455X (gfx1250) — matrix operand order repaired and checked by reading, not compiled, not run on hardware
//
#include <hip/hip_runtime.h>
#include <hip/hip_bf16.h>
#include <math.h>

// ---------------------------------------------------------------------------
// Types for WMMA
// ---------------------------------------------------------------------------
typedef __attribute__((ext_vector_type(16))) __bf16          v16bf;
typedef __attribute__((ext_vector_type(8)))  float           v8f;
typedef __attribute__((ext_vector_type(8)))  unsigned short  us8;

#define NNODES   100000
#define EDGES    1600000
#define DIN      256
#define DOUT     128
#define NCOLS    640           // 128 (w) + 256 (w1) + 256 (w2)
#define NTILES_N 40            // 640 / 16
#define KSTEPS   8             // 256 / 32
#define MROWS    32            // rows per block (two 16-row M-tiles)
#define EPS      1e-3f
#define LEAKY    0.2f

// float -> bf16 with round-to-nearest-even
__device__ __forceinline__ unsigned short f2bf(float f) {
    unsigned int u = __float_as_uint(f);
    unsigned int r = ((u >> 16) & 1u) + 0x7FFFu;
    return (unsigned short)((u + r) >> 16);
}
// bf16 -> float (exact)
__device__ __forceinline__ float bf2f(unsigned short h) {
    return __uint_as_float(((unsigned int)h) << 16);
}

// ---------------------------------------------------------------------------
// K0: zero the 512-float stats accumulators
// ---------------------------------------------------------------------------
__global__ void zero_stats(float* p) {
    int i = blockIdx.x * blockDim.x + threadIdx.x;
    if (i < 512) p[i] = 0.0f;
}

// ---------------------------------------------------------------------------
// K1: per-feature sum and sum-of-squares (coalesced: thread t owns column t)
// ---------------------------------------------------------------------------
__global__ __launch_bounds__(256) void bn_stats(const float* __restrict__ x,
                                                float* __restrict__ sums,
                                                float* __restrict__ sqs) {
    const int c   = threadIdx.x;                 // 0..255 == feature
    const int rpb = (NNODES + gridDim.x - 1) / gridDim.x;
    const int r0  = blockIdx.x * rpb;
    const int r1  = min(NNODES, r0 + rpb);
    float s = 0.0f, q = 0.0f;
    for (int r = r0; r < r1; ++r) {
        float v = x[(size_t)r * DIN + c];
        s += v;
        q += v * v;
    }
    atomicAdd(&sums[c], s);
    atomicAdd(&sqs[c],  q);
}

// ---------------------------------------------------------------------------
// K2: mean / rsqrt(var + eps)
// ---------------------------------------------------------------------------
__global__ void bn_finalize(const float* __restrict__ sums,
                            const float* __restrict__ sqs,
                            float* __restrict__ mean,
                            float* __restrict__ rsig) {
    int c = threadIdx.x;
    float m = sums[c] * (1.0f / (float)NNODES);
    float v = sqs[c]  * (1.0f / (float)NNODES) - m * m;
    mean[c] = m;
    rsig[c] = rsqrtf(v + EPS);
}

// ---------------------------------------------------------------------------
// K3: pack [w | w1 | w2] into fragment-major bf16 B buffer.
// Layout: Bp[((ntile*8 + kstep)*32 + lane)*16 + i]
//   lane: n = lane&15 ; K = kstep*32 + (lane>>4)*8 + (i&7) + (i>>3)*16   (B-matrix VGPR layout)
// ---------------------------------------------------------------------------
__global__ void pack_w(const float* __restrict__ w,
                       const float* __restrict__ w1,
                       const float* __restrict__ w2,
                       unsigned short* __restrict__ Bp) {
    int tid = blockIdx.x * blockDim.x + threadIdx.x;
    if (tid >= DIN * NCOLS) return;
    int i     = tid & 15;
    int lane  = (tid >> 4) & 31;
    int blk   = tid >> 9;            // ntile*8 + kstep
    int kstep = blk & 7;
    int ntile = blk >> 3;
    int k     = kstep * 32 + ((lane >> 4) * 8) + (i & 7) + ((i >> 3) * 16);
    int col   = ntile * 16 + (lane & 15);
    float v;
    if (col < 128)      v = w [k * 128 +  col       ];
    else if (col < 384) v = w1[k * 256 + (col - 128)];
    else                v = w2[k * 256 + (col - 384)];
    Bp[tid] = f2bf(v);
}

// ---------------------------------------------------------------------------
// K4: WMMA GEMM  Y = xn @ [w|w1|w2]  with fused epilogues:
//   cols   0..127 -> mapped (fp32)
//   cols 128..383 -> s1[i] += sum_j Y[i,j]*xn[i,j]   (then tanh)
//   cols 384..639 -> s2 likewise
// Block = 256 threads = 8 waves; TWO 16-row M-tiles per block so every
// B fragment load feeds two WMMAs (halves B L2 traffic, doubles math/load).
// ---------------------------------------------------------------------------
__global__ __launch_bounds__(256) void gemm_scores(
        const float* __restrict__ x,
        const unsigned short* __restrict__ Bp,
        const float* __restrict__ mean,
        const float* __restrict__ rsig,
        float* __restrict__ mapped,
        float* __restrict__ s1,
        float* __restrict__ s2) {
    __shared__ unsigned short At[MROWS][264]; // padded: 132 dwords/row -> conflict-free
    __shared__ float sacc[2][MROWS];

    const int tid  = threadIdx.x;
    const int lane = tid & 31;
    const int wave = tid >> 5;
    const long mbase = (long)blockIdx.x * MROWS;

    if (tid < 64) sacc[tid >> 5][tid & 31] = 0.0f;

    // cooperative load of the x tile -> normalize -> bf16 LDS (vectorized)
    // thread t handles rows (t>>4) and (t>>4)+16, 16 halves at col (t&15)*16
    {
        const int c0 = (tid & 15) * 16;     // 0..240
        const float4* mn4 = (const float4*)(mean + c0);
        const float4* rs4 = (const float4*)(rsig + c0);
        float4 mv[4], rv[4];
        #pragma unroll
        for (int q = 0; q < 4; ++q) { mv[q] = mn4[q]; rv[q] = rs4[q]; }
        #pragma unroll
        for (int half = 0; half < 2; ++half) {
            const int r = (tid >> 4) + half * 16;   // 0..31
            const float4* xr4 = (const float4*)(x + (mbase + r) * DIN + c0);
            unsigned short hv[16];
            #pragma unroll
            for (int q = 0; q < 4; ++q) {
                float4 xv = xr4[q];
                hv[q * 4 + 0] = f2bf((xv.x - mv[q].x) * rv[q].x);
                hv[q * 4 + 1] = f2bf((xv.y - mv[q].y) * rv[q].y);
                hv[q * 4 + 2] = f2bf((xv.z - mv[q].z) * rv[q].z);
                hv[q * 4 + 3] = f2bf((xv.w - mv[q].w) * rv[q].w);
            }
            us8* dst = (us8*)&At[r][c0];
            dst[0] = *(const us8*)&hv[0];
            dst[1] = *(const us8*)&hv[8];
        }
    }
    __syncthreads();

    const int m    = lane & 15;
    const int ksel = (lane >> 4) * 8;       // 16-bit A-matrix layout
    const int n    = lane & 15;
    const int rsel = (lane >> 4) * 8;       // C/D layout: VGPR r -> row r or 8+r

    const v8f vzero = {0.f, 0.f, 0.f, 0.f, 0.f, 0.f, 0.f, 0.f};
    v8f acc0[5] = {vzero, vzero, vzero, vzero, vzero};   // M-tile 0 (rows 0..15)
    v8f acc1[5] = {vzero, vzero, vzero, vzero, vzero};   // M-tile 1 (rows 16..31)

    union FU { us8 h[2]; v16bf v; };

    for (int ks = 0; ks < KSTEPS; ++ks) {
        FU A0, A1;
        A0.h[0] = *(const us8*)&At[m][ks * 32 + ksel];
        A0.h[1] = *(const us8*)&At[m][ks * 32 + ksel + 16];
        A1.h[0] = *(const us8*)&At[m + 16][ks * 32 + ksel];
        A1.h[1] = *(const us8*)&At[m + 16][ks * 32 + ksel + 16];
        #pragma unroll
        for (int t = 0; t < 5; ++t) {
            const int ntile = wave + 8 * t;
            FU B;
            const us8* bp = (const us8*)(Bp + ((size_t)(ntile * 8 + ks) * 32 + lane) * 16);
            B.h[0] = bp[0];
            B.h[1] = bp[1];
            acc0[t] = __builtin_amdgcn_wmma_f32_16x16x32_bf16(
                false, A0.v, false, B.v, (short)0, acc0[t], false, false);
            acc1[t] = __builtin_amdgcn_wmma_f32_16x16x32_bf16(
                false, A1.v, false, B.v, (short)0, acc1[t], false, false);
        }
    }

    #pragma unroll
    for (int t = 0; t < 5; ++t) {
        const int ntile = wave + 8 * t;     // wave-uniform
        #pragma unroll
        for (int mt = 0; mt < 2; ++mt) {
            const v8f* accp = mt ? acc1 : acc0;
            const int  lrow = mt * 16 + rsel;           // local row base in tile
            if (ntile < 8) {
                // mapped tile store
                const int col = ntile * 16 + n;
                #pragma unroll
                for (int r = 0; r < 8; ++r) {
                    long row = mbase + lrow + r;
                    mapped[row * DOUT + col] = accp[t][r];
                }
            } else {
                // quadratic-form partials: sum_j Y[i,j] * xn[i,j]
                const int which = (ntile < 24) ? 0 : 1;
                const int js    = (ntile - (which ? 24 : 8)) * 16 + n;
                float partial[8];
                #pragma unroll
                for (int r = 0; r < 8; ++r) {
                    float xv = bf2f(At[lrow + r][js]);
                    partial[r] = accp[t][r] * xv;
                }
                // reduce across the 16 columns (lanes differing in bits 0..3)
                #pragma unroll
                for (int mask = 8; mask; mask >>= 1) {
                    #pragma unroll
                    for (int r = 0; r < 8; ++r)
                        partial[r] += __shfl_xor(partial[r], mask, 32);
                }
                if ((lane & 15) == 0) {
                    #pragma unroll
                    for (int r = 0; r < 8; ++r)
                        atomicAdd(&sacc[which][lrow + r], partial[r]);
                }
            }
        }
    }
    __syncthreads();
    if (tid < 32)       s1[mbase + tid]        = tanhf(sacc[0][tid]);
    else if (tid < 64)  s2[mbase + (tid - 32)] = tanhf(sacc[1][tid - 32]);
}

// ---------------------------------------------------------------------------
// K5: CSR row pointers via binary search (rows[] is sorted)
// ---------------------------------------------------------------------------
__global__ void build_rowptr(const int* __restrict__ rows, int* __restrict__ rp) {
    int i = blockIdx.x * blockDim.x + threadIdx.x;
    if (i > NNODES) return;
    if (i == NNODES) { rp[NNODES] = EDGES; return; }
    int lo = 0, hi = EDGES;
    while (lo < hi) {
        int mid = (lo + hi) >> 1;
        if (rows[mid] < i) lo = mid + 1; else hi = mid;
    }
    rp[i] = lo;
}

// ---------------------------------------------------------------------------
// K6: per-row edge softmax (one wave per row, shfl reductions, no atomics)
// ---------------------------------------------------------------------------
__global__ __launch_bounds__(256) void edge_softmax(
        const int* __restrict__ cols,
        const float* __restrict__ edge_vals,
        const float* __restrict__ s1,
        const float* __restrict__ s2,
        const int* __restrict__ rp,
        float* __restrict__ alpha) {
    const int wid  = (blockIdx.x * blockDim.x + threadIdx.x) >> 5;
    const int lane = threadIdx.x & 31;
    if (wid >= NNODES) return;
    const int start = rp[wid], end = rp[wid + 1];
    if (start >= end) return;
    const float a = s1[wid];

    float mx = -INFINITY;
    for (int e = start + lane; e < end; e += 32) {
        float ev = edge_vals[e] * (a + s2[cols[e]]);
        ev = ev > 0.0f ? ev : LEAKY * ev;
        alpha[e] = ev;
        mx = fmaxf(mx, ev);
    }
    #pragma unroll
    for (int msk = 16; msk; msk >>= 1) mx = fmaxf(mx, __shfl_xor(mx, msk, 32));

    float sum = 0.0f;
    for (int e = start + lane; e < end; e += 32) {
        float ex = __expf(alpha[e] - mx);
        alpha[e] = ex;
        sum += ex;
    }
    #pragma unroll
    for (int msk = 16; msk; msk >>= 1) sum += __shfl_xor(sum, msk, 32);

    const float inv = 1.0f / sum;
    for (int e = start + lane; e < end; e += 32) alpha[e] *= inv;
}

// ---------------------------------------------------------------------------
// K7: SpMM  out[i,:] = sum_e alpha[e] * mapped[cols[e],:]
// One wave per row; lane owns 4 columns (float4). Writes every row.
// ---------------------------------------------------------------------------
__global__ __launch_bounds__(256) void spmm(
        const int* __restrict__ cols,
        const float* __restrict__ alpha,
        const float* __restrict__ mapped,
        const int* __restrict__ rp,
        float* __restrict__ out) {
    const int wid  = (blockIdx.x * blockDim.x + threadIdx.x) >> 5;
    const int lane = threadIdx.x & 31;
    if (wid >= NNODES) return;
    const int start = rp[wid], end = rp[wid + 1];
    const float4* mp = (const float4*)mapped;
    float4 acc = {0.f, 0.f, 0.f, 0.f};
    for (int e = start; e < end; ++e) {
        const float a = alpha[e];
        const int   c = cols[e];
        float4 v = mp[(size_t)c * 32 + lane];
        acc.x += a * v.x; acc.y += a * v.y; acc.z += a * v.z; acc.w += a * v.w;
    }
    ((float4*)out)[(size_t)wid * 32 + lane] = acc;
}

// ---------------------------------------------------------------------------
// Launcher
// ---------------------------------------------------------------------------
extern "C" void kernel_launch(void* const* d_in, const int* in_sizes, int n_in,
                              void* d_out, int out_size, void* d_ws, size_t ws_size,
                              hipStream_t stream) {
    const float* x         = (const float*)d_in[0];
    const float* w         = (const float*)d_in[1];
    const float* w1        = (const float*)d_in[2];
    const float* w2        = (const float*)d_in[3];
    const float* edge_vals = (const float*)d_in[4];
    const int*   rows      = (const int*)d_in[5];
    const int*   cols      = (const int*)d_in[6];
    float*       out       = (float*)d_out;

    // workspace carve-up (16B aligned)
    char* ws = (char*)d_ws;
    size_t off = 0;
    auto take = [&](size_t bytes) { char* p = ws + off; off = (off + bytes + 15) & ~(size_t)15; return p; };
    float*          sums    = (float*)take(256 * 4);
    float*          sqs     = (float*)take(256 * 4);
    float*          mean    = (float*)take(256 * 4);
    float*          rsig    = (float*)take(256 * 4);
    unsigned short* Bp      = (unsigned short*)take((size_t)DIN * NCOLS * 2);
    float*          mapped  = (float*)take((size_t)NNODES * DOUT * 4);
    float*          s1      = (float*)take((size_t)NNODES * 4);
    float*          s2      = (float*)take((size_t)NNODES * 4);
    int*            rp      = (int*)take((size_t)(NNODES + 1) * 4);
    float*          alpha   = (float*)take((size_t)EDGES * 4);
    (void)ws_size; (void)in_sizes; (void)n_in; (void)out_size;

    // K0: zero stats accumulators (sums+sqs are contiguous: 512 floats)
    zero_stats<<<2, 256, 0, stream>>>(sums);
    // K1: column stats
    bn_stats<<<400, 256, 0, stream>>>(x, sums, sqs);
    // K2: finalize mean/rsig
    bn_finalize<<<1, 256, 0, stream>>>(sums, sqs, mean, rsig);
    // K3: pack weights to fragment-major bf16
    pack_w<<<(DIN * NCOLS + 255) / 256, 256, 0, stream>>>(w, w1, w2, Bp);
    // K4: WMMA GEMM + fused score epilogue  (100000/32 = 3125 blocks)
    gemm_scores<<<NNODES / MROWS, 256, 0, stream>>>(x, Bp, mean, rsig, mapped, s1, s2);
    // K5: CSR row pointers
    build_rowptr<<<(NNODES + 1 + 255) / 256, 256, 0, stream>>>(rows, rp);
    // K6: edge softmax (one wave per row -> 8 rows per 256-thread block)
    edge_softmax<<<(NNODES + 7) / 8, 256, 0, stream>>>(cols, edge_vals, s1, s2, rp, alpha);
    // K7: SpMM
    spmm<<<(NNODES + 7) / 8, 256, 0, stream>>>(cols, alpha, mapped, rp, out);
}
